// FeatureCorrelation_14207751815756
// MI455X (gfx1250) — matrix operand order repaired and checked by reading, not compiled, not run on hardware
//
#include <hip/hip_runtime.h>
#include <hip/hip_bf16.h>

typedef __attribute__((ext_vector_type(8)))  float        v8f;
typedef __attribute__((ext_vector_type(16))) __bf16       v16bf;
typedef __attribute__((ext_vector_type(4)))  unsigned int v4u;

union Op32 { v16bf bf; v4u u[2]; };   // 32 bytes: one WMMA bf16 operand half-set

#define TEMP_INV 14.2857142857142857f   // 1 / 0.07

static constexpr int NB = 32;    // batch
static constexpr int NC = 256;   // channels (K)
static constexpr int HW = 1024;  // 32*32 positions

// ---------------------------------------------------------------------------
// Stage 0: split f32 -> (hi, lo) bf16 with RNE, and transpose [c][pos] ->
// [pos][c] so GEMM operands load as contiguous-K global_load_b128.
// ---------------------------------------------------------------------------
__device__ __forceinline__ unsigned int bf16_rne_bits(float x) {
    unsigned int u = __float_as_uint(x);
    return (u + 0x7FFFu + ((u >> 16) & 1u)) >> 16;
}

__global__ __launch_bounds__(256) void split_transpose_kernel(
    const float* __restrict__ in,            // [b][c][pos]
    unsigned short* __restrict__ hi,         // [b][pos][c]
    unsigned short* __restrict__ lo)         // [b][pos][c]
{
    __shared__ unsigned short hT[32][33];
    __shared__ unsigned short lT[32][33];
    const int b  = blockIdx.z;
    const int p0 = blockIdx.x * 32;
    const int c0 = blockIdx.y * 32;
    const int t  = threadIdx.x;

    const float* src = in + ((size_t)b * NC + c0) * HW + p0;
    const int pos = t & 31;
    const int cr  = t >> 5;
#pragma unroll
    for (int q = 0; q < 4; ++q) {
        const int cc = cr + 8 * q;
        const float x = src[(size_t)cc * HW + pos];
        const unsigned int hb = bf16_rne_bits(x);
        const float hf = __uint_as_float(hb << 16);
        const unsigned int lb = bf16_rne_bits(x - hf);
        hT[cc][pos] = (unsigned short)hb;
        lT[cc][pos] = (unsigned short)lb;
    }
    __syncthreads();

    unsigned short* dh = hi + ((size_t)b * HW + p0) * NC + c0;
    unsigned short* dl = lo + ((size_t)b * HW + p0) * NC + c0;
    const int c2 = t & 31;
    const int pr = t >> 5;
#pragma unroll
    for (int q = 0; q < 4; ++q) {
        const int pp = pr + 8 * q;
        dh[(size_t)pp * NC + c2] = hT[c2][pp];
        dl[(size_t)pp * NC + c2] = lT[c2][pp];
    }
}

// ---------------------------------------------------------------------------
// Stage 1: M2[b][i][jm] = (1/T) * sum_c fB[c][i] * fA[c][jm], via the
// error-compensated bf16 split:  M ~= Ah*Bh + Ah*Bl + Al*Bh  with f32 acc.
// One wave computes a 16(i) x 64(j) strip; all operand loads are b128.
// ---------------------------------------------------------------------------
__global__ __launch_bounds__(128) void gemm_bf16split_kernel(
    const unsigned short* __restrict__ rhi, const unsigned short* __restrict__ rlo,  // feature_B staged
    const unsigned short* __restrict__ chi, const unsigned short* __restrict__ clo,  // feature_A staged
    float* __restrict__ M2)
{
    const int lane = threadIdx.x & 31;
    const int wave = threadIdx.x >> 5;
    const int b    = blockIdx.z;
    const int i0   = (blockIdx.y * 4 + wave) * 16;
    const int j0   = blockIdx.x * 64;

    const int m  = lane & 15;
    const int ak = (lane >> 4) << 3;   // A-operand K sub-base: 0 / 8
    const int bk = (lane >> 4) << 3;   // B-operand K sub-base: 0 / 8

    const unsigned short* rh = rhi + ((size_t)b * HW + i0 + m) * NC + ak;
    const unsigned short* rl = rlo + ((size_t)b * HW + i0 + m) * NC + ak;
    const unsigned short* ch = chi + ((size_t)b * HW + j0 + m) * NC + bk;
    const unsigned short* cl = clo + ((size_t)b * HW + j0 + m) * NC + bk;

    v8f acc[4] = {{}, {}, {}, {}};

    for (int k0 = 0; k0 < NC; k0 += 32) {
        Op32 ah, al;
        ah.u[0] = *(const v4u*)(rh + k0);        // VGPR0-3: K = ak..ak+7
        ah.u[1] = *(const v4u*)(rh + k0 + 16);   // VGPR4-7: K = ak+16..ak+23
        al.u[0] = *(const v4u*)(rl + k0);
        al.u[1] = *(const v4u*)(rl + k0 + 16);
#pragma unroll
        for (int t = 0; t < 4; ++t) {
            const unsigned short* ph = ch + (size_t)(16 * t) * NC + k0;
            const unsigned short* pl = cl + (size_t)(16 * t) * NC + k0;
            Op32 bh, bl;
            bh.u[0] = *(const v4u*)(ph);         // VGPR0-3: K = bk..bk+7
            bh.u[1] = *(const v4u*)(ph + 16);    // VGPR4-7: K = bk+16..bk+23
            bl.u[0] = *(const v4u*)(pl);
            bl.u[1] = *(const v4u*)(pl + 16);
            acc[t] = __builtin_amdgcn_wmma_f32_16x16x32_bf16(false, ah.bf, false, bh.bf, (short)0, acc[t], false, false);
            acc[t] = __builtin_amdgcn_wmma_f32_16x16x32_bf16(false, ah.bf, false, bl.bf, (short)0, acc[t], false, false);
            acc[t] = __builtin_amdgcn_wmma_f32_16x16x32_bf16(false, al.bf, false, bh.bf, (short)0, acc[t], false, false);
        }
    }

    // C/D layout: row = i0 + v + 8*(lane>>4), col = lane&15 per 16-wide tile
    float* __restrict__ out = M2 + ((size_t)b << 20);
    const int rbase = i0 + ((lane >> 4) << 3);
    const int n     = lane & 15;
#pragma unroll
    for (int v = 0; v < 8; ++v) {
        const size_t r = (size_t)(rbase + v) * HW;
        out[r + j0 +  0 + n] = acc[0][v] * TEMP_INV;
        out[r + j0 + 16 + n] = acc[1][v] * TEMP_INV;
        out[r + j0 + 32 + n] = acc[2][v] * TEMP_INV;
        out[r + j0 + 48 + n] = acc[3][v] * TEMP_INV;
    }
}

// ---------------------------------------------------------------------------
// Stage 2a: per-row (over j) max and exp-sum. One 256-thread block per row.
// ---------------------------------------------------------------------------
__global__ __launch_bounds__(256) void row_stats_kernel(
    const float* __restrict__ M2, float* __restrict__ rmax, float* __restrict__ rsum)
{
    __shared__ float red[256];
    const int   row = blockIdx.x;            // b*1024 + i
    const int   t   = threadIdx.x;
    const float* p  = M2 + (size_t)row * HW;

    float x0 = p[t], x1 = p[t + 256], x2 = p[t + 512], x3 = p[t + 768];
    float mx = fmaxf(fmaxf(x0, x1), fmaxf(x2, x3));
    red[t] = mx;
    __syncthreads();
    for (int s = 128; s > 0; s >>= 1) {
        if (t < s) red[t] = fmaxf(red[t], red[t + s]);
        __syncthreads();
    }
    mx = red[0];
    __syncthreads();

    float sm = __expf(x0 - mx) + __expf(x1 - mx) + __expf(x2 - mx) + __expf(x3 - mx);
    red[t] = sm;
    __syncthreads();
    for (int s = 128; s > 0; s >>= 1) {
        if (t < s) red[t] += red[t + s];
        __syncthreads();
    }
    if (t == 0) { rmax[row] = mx; rsum[row] = red[0]; }
}

// ---------------------------------------------------------------------------
// Stage 2b: per-column (over i) online max/exp-sum. Coalesced column reads.
// ---------------------------------------------------------------------------
__global__ __launch_bounds__(256) void col_stats_kernel(
    const float* __restrict__ M2, float* __restrict__ cmax, float* __restrict__ csum)
{
    const int b = blockIdx.y;
    const int j = blockIdx.x * 256 + threadIdx.x;
    const float* p = M2 + ((size_t)b << 20) + j;

    float mx = -3.402823466e38f, sm = 0.0f;
    for (int i = 0; i < HW; ++i) {
        float x  = p[(size_t)i * HW];
        float nm = fmaxf(mx, x);
        sm = sm * __expf(mx - nm) + __expf(x - nm);
        mx = nm;
    }
    cmax[b * HW + j] = mx;
    csum[b * HW + j] = sm;
}

// ---------------------------------------------------------------------------
// Stage 3: P = exp(2x - rmax - cmax)/(rsum*csum); 32x32 LDS tile transpose;
// output row index applies the (w,h) permutation of A's flattening.
// ---------------------------------------------------------------------------
__global__ __launch_bounds__(256) void output_kernel(
    const float* __restrict__ M2,
    const float* __restrict__ rmax, const float* __restrict__ rsum,
    const float* __restrict__ cmax, const float* __restrict__ csum,
    float* __restrict__ out)
{
    __shared__ float tile[32][33];
    const int b  = blockIdx.z;
    const int i0 = blockIdx.y * 32;
    const int j0 = blockIdx.x * 32;

    const float* __restrict__ Mb = M2 + ((size_t)b << 20);
    const float* rm = rmax + b * HW;
    const float* rs = rsum + b * HW;
    const float* cm = cmax + b * HW;
    const float* cs = csum + b * HW;

    const int c = threadIdx.x & 31;
    const int r = threadIdx.x >> 5;   // 0..7

    const float cmx  = cm[j0 + c];
    const float cinv = 1.0f / cs[j0 + c];

#pragma unroll
    for (int q = 0; q < 4; ++q) {
        const int rr = r + 8 * q;
        const float x = Mb[(size_t)(i0 + rr) * HW + j0 + c];
        tile[rr][c] = __expf(2.0f * x - rm[i0 + rr] - cmx) * (cinv / rs[i0 + rr]);
    }
    __syncthreads();

#pragma unroll
    for (int q = 0; q < 4; ++q) {
        const int cc   = (threadIdx.x >> 5) + 8 * q;   // jm within tile
        const int rp   = threadIdx.x & 31;             // i within tile
        const int jm   = j0 + cc;
        const int jout = ((jm & 31) << 5) | (jm >> 5); // inverse (w,h) permutation
        out[((size_t)b * HW + jout) * HW + (i0 + rp)] = tile[rp][cc];
    }
}

// ---------------------------------------------------------------------------
extern "C" void kernel_launch(void* const* d_in, const int* in_sizes, int n_in,
                              void* d_out, int out_size, void* d_ws, size_t ws_size,
                              hipStream_t stream)
{
    const float* fA = (const float*)d_in[0];   // feature_A (32,256,32,32)
    const float* fB = (const float*)d_in[1];   // feature_B (32,256,32,32)
    float* outp = (float*)d_out;               // (32,1024,32,32)

    // Workspace layout (all offsets 16B-aligned):
    //   M2   : 32*1024*1024 f32            (134.2 MB)
    //   stats: 4 * 32*1024 f32             (0.5 MB)
    //   bf16 staging: 4 * 32*1024*256 u16  (64 MB)
    float* M2   = (float*)d_ws;
    const size_t mElems = (size_t)NB * HW * HW;            // 33,554,432
    float* rmax = M2 + mElems;
    float* rsum = rmax + (size_t)NB * HW;
    float* cmax = rsum + (size_t)NB * HW;
    float* csum = cmax + (size_t)NB * HW;

    unsigned short* staged = (unsigned short*)(csum + (size_t)NB * HW);
    const size_t sElems = (size_t)NB * HW * NC;            // 8,388,608 per array
    unsigned short* rhi = staged;                // feature_B hi  [b][i][c]
    unsigned short* rlo = rhi + sElems;          // feature_B lo
    unsigned short* chi = rlo + sElems;          // feature_A hi  [b][jm][c]
    unsigned short* clo = chi + sElems;          // feature_A lo

    // Stage 0: split + transpose both inputs to K-contiguous bf16 hi/lo
    split_transpose_kernel<<<dim3(HW / 32, NC / 32, NB), 256, 0, stream>>>(fB, rhi, rlo);
    split_transpose_kernel<<<dim3(HW / 32, NC / 32, NB), 256, 0, stream>>>(fA, chi, clo);

    // Stage 1: batched split-bf16 WMMA GEMM (3 terms, f32 accumulate)
    gemm_bf16split_kernel<<<dim3(HW / 64, HW / (16 * 4), NB), 128, 0, stream>>>(
        rhi, rlo, chi, clo, M2);

    // Stage 2: softmax statistics
    row_stats_kernel<<<dim3(NB * HW), 256, 0, stream>>>(M2, rmax, rsum);
    col_stats_kernel<<<dim3(HW / 256, NB), 256, 0, stream>>>(M2, cmax, csum);

    // Stage 3: fused double-softmax product + permuted transpose store
    output_kernel<<<dim3(HW / 32, HW / 32, NB), 256, 0, stream>>>(
        M2, rmax, rsum, cmax, csum, outp);
}
